// VQGANCodebook_44538810860102
// MI455X (gfx1250) — matrix operand order repaired and checked by reading, not compiled, not run on hardware
//
#include <hip/hip_runtime.h>
#include <hip/hip_bf16.h>

// ---------------------------------------------------------------------------
// VQ-VAE / VQGAN codebook quantization for MI455X (gfx1250, wave32, WMMA)
//
// Shapes: z_e [32,256,32,32] f32 (NCHW), embedding [1024,256] f32
//   N = 32*32*32 = 32768 rows, D = 256, K = 1024 codes
// Outputs flat: loss(1) | z_q_out(8388608) | perplexity(1) | encodings(33554432)
// ---------------------------------------------------------------------------

#define NROWS      32768
#define DIM        256
#define KCODES     1024
#define ZQ_ELEMS   8388608   // 32*256*32*32
#define OUT_ZQ_OFF 1
#define OUT_PP_OFF 8388609
#define OUT_ENC_OFF 8388610

typedef __attribute__((ext_vector_type(16))) __bf16 v16bf;
typedef __attribute__((ext_vector_type(8)))  float  v8f;
typedef __attribute__((ext_vector_type(2)))  float  vf2;   // ext-vector for NT stores

static __device__ __forceinline__ __bf16 f2bf(float f) {
  // round-to-nearest-even f32 -> bf16 (inputs are finite)
  union { float f; unsigned u; } x; x.f = f;
  unsigned r = (x.u + 0x7fffu + ((x.u >> 16) & 1u)) >> 16;
  union { unsigned short s; __bf16 b; } y; y.s = (unsigned short)r;
  return y.b;
}

// ---------------------------------------------------------------------------
// Kernel 1: embedding f32->bf16 pack, e_sq[k] = ||e_k||^2, zero counts.
// One block (256 threads) per code.
// ---------------------------------------------------------------------------
__global__ __launch_bounds__(256) void vq_prep_kernel(
    const float* __restrict__ emb, unsigned short* __restrict__ ebf,
    float* __restrict__ esq, int* __restrict__ counts) {
  __shared__ float red[256];
  int k = blockIdx.x, t = threadIdx.x;
  float v = emb[k * DIM + t];
  union { __bf16 b; unsigned short s; } cv; cv.b = f2bf(v);
  ebf[k * DIM + t] = cv.s;
  red[t] = v * v;
  __syncthreads();
  #pragma unroll
  for (int o = 128; o > 0; o >>= 1) {
    if (t < o) red[t] += red[t + o];
    __syncthreads();
  }
  if (t == 0) { esq[k] = red[0]; counts[k] = 0; }
}

// ---------------------------------------------------------------------------
// Kernel 2: argmin over codes via bf16 WMMA, M-blocked.
// Each wave32 owns 32 rows of z (two A tiles 16x256 bf16 in registers); every
// B v16bf load feeds two WMMAs -> halves L2 B-traffic vs 16-row tiling.
// argmin_n( e_sq[n] - 2 * <z_m, e_n> )   (z_sq is row-constant -> dropped)
// ---------------------------------------------------------------------------
__global__ __launch_bounds__(128) void vq_argmin_kernel(
    const float* __restrict__ z_e, const unsigned short* __restrict__ ebf_u16,
    const float* __restrict__ esq, int* __restrict__ idx_out) {
  const int lane = threadIdx.x & 31;
  const int wave = threadIdx.x >> 5;
  const int wid  = blockIdx.x * 4 + wave;        // 0..1023, 32 rows each
  const int hi   = lane >> 4;                    // lane half (A/B K split)
  const int m    = lane & 15;                    // A row / B column in tile

  // Rows nrow0 = wid*32 + m (tile0) and nrow0+16 (tile1) share b,h; w = m / m+16.
  // NCHW source: n = b*1024 + h*32 + w ; z[((b*256+k)*32+h)*32+w]
  const int nrow = wid * 32 + m;
  const int b = nrow >> 10, h = (nrow >> 5) & 31;
  const int zbase = b * 262144 + h * 32 + m;     // + k*1024 per element

  // Load both A tiles once, in contiguous K order (not the ISA table order):
  // lane half hi: element i -> K = 16*hi + i (contiguous, matches B)
  v16bf a0[8], a1[8];
  #pragma unroll
  for (int s = 0; s < 8; ++s) {
    #pragma unroll
    for (int i = 0; i < 16; ++i) {
      const int j = i >> 1, lo = i & 1;
      const int k = s * 32 + hi * 16 + i;
      a0[s][i] = f2bf(z_e[zbase + k * 1024]);
      a1[s][i] = f2bf(z_e[zbase + 16 + k * 1024]);
    }
  }

  float minv0[8], minv1[8];
  int   mini0[8], mini1[8];
  #pragma unroll
  for (int r = 0; r < 8; ++r) {
    minv0[r] = 3.4e38f; mini0[r] = 0;
    minv1[r] = 3.4e38f; mini1[r] = 0;
  }

  const __bf16* __restrict__ ebf = (const __bf16*)ebf_u16;

  #pragma unroll 2
  for (int chunk = 0; chunk < 64; ++chunk) {
    const int col = chunk * 16 + m;              // this lane's B column
    v8f acc0 = {}, acc1 = {};
    #pragma unroll
    for (int s = 0; s < 8; ++s) {
      // B layout: lane half hi, column m, K = hi*16 + 2v + lo == contiguous
      // 16 bf16 from row-major embedding -> single 32B v16bf load, used twice.
      v16bf bv = *(const v16bf*)(ebf + (size_t)col * DIM + s * 32 + hi * 16);
      acc0 = __builtin_amdgcn_wmma_f32_16x16x32_bf16(
          false, a0[s], false, bv, (short)0, acc0, false, false);
      acc1 = __builtin_amdgcn_wmma_f32_16x16x32_bf16(
          false, a1[s], false, bv, (short)0, acc1, false, false);
    }
    const float e2 = esq[col];
    #pragma unroll
    for (int r = 0; r < 8; ++r) {                // rows m = hi*8 + r per tile
      const float d0 = e2 - 2.0f * acc0[r];
      if (d0 < minv0[r]) { minv0[r] = d0; mini0[r] = col; }
      const float d1 = e2 - 2.0f * acc1[r];
      if (d1 < minv1[r]) { minv1[r] = d1; mini1[r] = col; }
    }
  }

  // Reduce (min, argmin) across the 16 lanes of each half (cols n mod 16);
  // tie-break to the lower index to match argmin semantics.
  #pragma unroll
  for (int r = 0; r < 8; ++r) {
    float v0 = minv0[r]; int i0 = mini0[r];
    float v1 = minv1[r]; int i1 = mini1[r];
    #pragma unroll
    for (int off = 8; off > 0; off >>= 1) {
      const float ov0 = __shfl_xor(v0, off, 32);
      const int   oi0 = __shfl_xor(i0, off, 32);
      if (ov0 < v0 || (ov0 == v0 && oi0 < i0)) { v0 = ov0; i0 = oi0; }
      const float ov1 = __shfl_xor(v1, off, 32);
      const int   oi1 = __shfl_xor(i1, off, 32);
      if (ov1 < v1 || (ov1 == v1 && oi1 < i1)) { v1 = ov1; i1 = oi1; }
    }
    if (m == r) {                                // two writers per r: hi = 0/1
      idx_out[wid * 32 + hi * 8 + r]      = i0;
      idx_out[wid * 32 + 16 + hi * 8 + r] = i1;
    }
  }
}

// ---------------------------------------------------------------------------
// Kernel 3: encodings one-hot. One block per row: zero 1024 floats, set 1.0,
// bump histogram counts. Non-temporal stores: 134 MB of pure streaming output
// that nothing re-reads -- keep it out of L2. enc base is 8-byte aligned
// (float offset 8388610).
// ---------------------------------------------------------------------------
__global__ __launch_bounds__(256) void vq_enc_kernel(
    const int* __restrict__ idxs, float* __restrict__ enc,
    int* __restrict__ counts) {
  const int n = blockIdx.x, t = threadIdx.x;
  float* row = enc + (size_t)n * KCODES;
  vf2 z2 = {0.f, 0.f};
  __builtin_nontemporal_store(z2, (vf2*)row + t);
  __builtin_nontemporal_store(z2, (vf2*)row + t + 256);
  __syncthreads();
  if (t == 0) {
    const int k = idxs[n];
    __builtin_nontemporal_store(1.0f, row + k);
    atomicAdd(&counts[k], 1);
  }
}

// ---------------------------------------------------------------------------
// Kernel 4: z_q gather (NCHW) + deterministic per-block SSE partials.
// z_q_st == z_q in value (straight-through), loss = 1.25 * mean((z_q - z)^2).
// ---------------------------------------------------------------------------
__global__ __launch_bounds__(256) void vq_zq_loss_kernel(
    const float* __restrict__ z_e, const float* __restrict__ emb,
    const int* __restrict__ idxs, float* __restrict__ zq_out,
    float* __restrict__ block_sums) {
  __shared__ float red[256];
  const int gid = blockIdx.x * 256 + threadIdx.x;   // 0 .. 8388607
  const int w = gid & 31, hh = (gid >> 5) & 31, c = (gid >> 10) & 255,
            bb = gid >> 18;
  const int n = (bb << 10) + (hh << 5) + w;
  const int k = idxs[n];
  const float q  = emb[k * DIM + c];
  const float zz = z_e[gid];
  zq_out[gid] = q;
  const float d = q - zz;
  red[threadIdx.x] = d * d;
  __syncthreads();
  #pragma unroll
  for (int o = 128; o > 0; o >>= 1) {
    if (threadIdx.x < o) red[threadIdx.x] += red[threadIdx.x + o];
    __syncthreads();
  }
  if (threadIdx.x == 0) block_sums[blockIdx.x] = red[0];
}

// ---------------------------------------------------------------------------
// Kernel 5: finalize loss + perplexity. Single block, fixed-order reduction
// (bitwise deterministic across replays).
// ---------------------------------------------------------------------------
__global__ __launch_bounds__(1024) void vq_finalize_kernel(
    const int* __restrict__ counts, const float* __restrict__ block_sums,
    float* __restrict__ out) {
  __shared__ float red[1024];
  const int t = threadIdx.x;
  float s = 0.f;
  for (int i = t; i < 32768; i += 1024) s += block_sums[i];
  red[t] = s;
  __syncthreads();
  #pragma unroll
  for (int o = 512; o > 0; o >>= 1) {
    if (t < o) red[t] += red[t + o];
    __syncthreads();
  }
  const float sse = red[0];
  __syncthreads();
  const float p = (float)counts[t] * (1.0f / (float)NROWS);
  red[t] = p * __logf(p + 1e-10f);
  __syncthreads();
  #pragma unroll
  for (int o = 512; o > 0; o >>= 1) {
    if (t < o) red[t] += red[t + o];
    __syncthreads();
  }
  if (t == 0) {
    out[0]          = 1.25f * sse * (1.0f / (float)ZQ_ELEMS);  // (1+beta)*mse
    out[OUT_PP_OFF] = __expf(-red[0]);                         // perplexity
  }
}

// ---------------------------------------------------------------------------
extern "C" void kernel_launch(void* const* d_in, const int* in_sizes, int n_in,
                              void* d_out, int out_size, void* d_ws,
                              size_t ws_size, hipStream_t stream) {
  const float* z_e = (const float*)d_in[0];   // 8,388,608 f32
  const float* emb = (const float*)d_in[1];   //   262,144 f32
  float* out = (float*)d_out;

  // workspace carve-up (bytes)
  char* ws = (char*)d_ws;
  unsigned short* ebf   = (unsigned short*)(ws + 0);        // 524,288
  float*          esq   = (float*)(ws + 524288);            //   4,096
  int*            idxs  = (int*)  (ws + 528384);            // 131,072
  int*            cnts  = (int*)  (ws + 659456);            //   4,096
  float*          bsums = (float*)(ws + 663552);            // 131,072

  vq_prep_kernel<<<KCODES, 256, 0, stream>>>(emb, ebf, esq, cnts);
  vq_argmin_kernel<<<NROWS / 128, 128, 0, stream>>>(z_e, ebf, esq, idxs);
  vq_enc_kernel<<<NROWS, 256, 0, stream>>>(idxs, out + OUT_ENC_OFF, cnts);
  vq_zq_loss_kernel<<<ZQ_ELEMS / 256, 256, 0, stream>>>(
      z_e, emb, idxs, out + OUT_ZQ_OFF, bsums);
  vq_finalize_kernel<<<1, 1024, 0, stream>>>(cnts, bsums, out);
}
